// BlockFFTDirectPrior_28269474742317
// MI455X (gfx1250) — matrix operand order repaired and checked by reading, not compiled, not run on hardware
//
#include <hip/hip_runtime.h>
#include <hip/hip_bf16.h>

typedef __attribute__((ext_vector_type(16))) __bf16        v16bf;
typedef __attribute__((ext_vector_type(8)))  float         v8f;
typedef __attribute__((ext_vector_type(4)))  unsigned int  u32x4;
typedef __attribute__((ext_vector_type(4)))  float         f32x4;
typedef unsigned short u16;

#define NB    4096   // batch rows of x
#define DIN   4096
#define DOUT  4096
#define BS    256
#define KIN   16
#define KOUT  16
#define KHALF 129

// ---------------------------------------------------------------------------
// Kernel 1: half-spectrum -> real time-domain taps  w[i][j][d], d in [0,256)
// ---------------------------------------------------------------------------
__global__ void k_make_w(const float* __restrict__ Wr, const float* __restrict__ Wi,
                         float* __restrict__ w) {
  const int ij = blockIdx.x;       // i*16 + j   (256 blocks)
  const int d  = threadIdx.x;      // 0..255
  const float* wr = Wr + (size_t)ij * KHALF;
  const float* wi = Wi + (size_t)ij * KHALF;
  float acc  = wr[0];
  const float step = 6.28318530717958647692f * (float)d * (1.0f / 256.0f);
  for (int n = 1; n < 128; ++n) {
    float s, c;
    __sincosf(step * (float)n, &s, &c);
    acc += 2.0f * (wr[n] * c - wi[n] * s);
  }
  acc += (d & 1) ? -wr[128] : wr[128];
  w[(size_t)ij * BS + d] = acc * (1.0f / 256.0f);
}

// ---------------------------------------------------------------------------
// Kernel 2: materialize bf16 transposed operator  AT[dout][din]
//   AT[i*256+t][j*256+s] = bf16( w[i][j][(s - t) & 255] )
// ---------------------------------------------------------------------------
__global__ void k_make_AT(const float* __restrict__ w, u16* __restrict__ AT) {
  const int dout = blockIdx.x;     // 0..4095
  const int j    = blockIdx.y;     // 0..15
  const int s    = threadIdx.x;    // 0..255
  const int i = dout >> 8;
  const int t = dout & 255;
  float v = w[((size_t)(i * KIN + j) * BS) + ((unsigned)(s - t) & 255u)];
  __hip_bfloat16 h = __float2bfloat16(v);
  AT[(size_t)dout * DIN + (size_t)j * BS + s] = *(u16*)&h;
}

// ---------------------------------------------------------------------------
// Kernel 3: x (f32) -> bf16, 8 elements/thread, 16B stores
// ---------------------------------------------------------------------------
__global__ void k_cvt_x(const float* __restrict__ x, u16* __restrict__ xb) {
  const size_t base = ((size_t)blockIdx.x * blockDim.x + threadIdx.x) * 8;
  f32x4 a = *(const f32x4*)(x + base);
  f32x4 b = *(const f32x4*)(x + base + 4);
  u16 tmp[8];
  float vals[8] = {a.x, a.y, a.z, a.w, b.x, b.y, b.z, b.w};
#pragma unroll
  for (int q = 0; q < 8; ++q) {
    __hip_bfloat16 h = __float2bfloat16(vals[q]);
    tmp[q] = *(u16*)&h;
  }
  *(u32x4*)(xb + base) = *(const u32x4*)tmp;
}

// ---------------------------------------------------------------------------
// Kernel 4: out(4096x4096 f32) = Xb @ AT^T via WMMA bf16, f32 accumulate.
//   WG: 256 threads = 8 waves. Block tile 128(M) x 256(N), K-step 32.
//   Wave grid 2(M) x 4(N): each wave owns 64x64 = 4x4 subtiles (128 acc VGPRs).
//   LDS double-buffered, register-staged global loads, ONE barrier per K-step.
// ---------------------------------------------------------------------------
#define BM 128
#define BN 256
#define BK 32
#define LSTR 40   // LDS row stride in u16 (80B: 16B-aligned, conflict-free frag reads)

__global__ void __launch_bounds__(256)
k_gemm(const u16* __restrict__ Xb, const u16* __restrict__ AT,
       float* __restrict__ out) {
  __shared__ __align__(16) u16 xs[2 * BM * LSTR];   // 20 KB
  __shared__ __align__(16) u16 as[2 * BN * LSTR];   // 40 KB

  const int tid    = threadIdx.x;
  const int lane   = tid & 31;
  const int wave   = tid >> 5;
  const int wm     = wave >> 2;        // 0..1
  const int wn     = wave & 3;         // 0..3
  const int laneHi = lane >> 4;        // 0 or 1
  const int lane15 = lane & 15;

  const int m0 = blockIdx.y * BM;
  const int n0 = blockIdx.x * BN;
  const u16* gx = Xb + (size_t)m0 * DIN;
  const u16* ga = AT + (size_t)n0 * DIN;

  // staging chunk coordinates (row-major, 8 bf16 per 16B chunk)
  const int xrow[2] = { (tid * 2 + 0) >> 2, (tid * 2 + 1) >> 2 };
  const int xcol[2] = { ((tid * 2 + 0) & 3) * 8, ((tid * 2 + 1) & 3) * 8 };
  int arow[4], acol[4];
#pragma unroll
  for (int q = 0; q < 4; ++q) { arow[q] = (tid * 4 + q) >> 2; acol[q] = ((tid * 4 + q) & 3) * 8; }

  v8f acc[4][4];
#pragma unroll
  for (int a = 0; a < 4; ++a)
#pragma unroll
    for (int b = 0; b < 4; ++b)
      acc[a][b] = (v8f){0.f, 0.f, 0.f, 0.f, 0.f, 0.f, 0.f, 0.f};

  u32x4 rx[2], ra[4];

  // ---- prologue: fetch tile 0 and park it in LDS buffer 0
#pragma unroll
  for (int q = 0; q < 2; ++q)
    rx[q] = *(const u32x4*)&gx[(size_t)xrow[q] * DIN + xcol[q]];
#pragma unroll
  for (int q = 0; q < 4; ++q)
    ra[q] = *(const u32x4*)&ga[(size_t)arow[q] * DIN + acol[q]];
#pragma unroll
  for (int q = 0; q < 2; ++q)
    *(u32x4*)&xs[xrow[q] * LSTR + xcol[q]] = rx[q];
#pragma unroll
  for (int q = 0; q < 4; ++q)
    *(u32x4*)&as[arow[q] * LSTR + acol[q]] = ra[q];
  __syncthreads();

  for (int i = 0; i < DIN / BK; ++i) {
    const int cur = i & 1;
    const u16* xsb = &xs[cur * BM * LSTR];
    const u16* asb = &as[cur * BN * LSTR];
    const int kn = (i + 1) * BK;

    // ---- issue global loads for the NEXT K-tile (latency hides under WMMA)
    if (kn < DIN) {
#pragma unroll
      for (int q = 0; q < 2; ++q)
        rx[q] = *(const u32x4*)&gx[(size_t)xrow[q] * DIN + kn + xcol[q]];
#pragma unroll
      for (int q = 0; q < 4; ++q)
        ra[q] = *(const u32x4*)&ga[(size_t)arow[q] * DIN + kn + acol[q]];
    }
    if ((i + 2) * BK < DIN) {   // warm caches two tiles ahead
      __builtin_prefetch(&gx[(size_t)xrow[0] * DIN + (i + 2) * BK + xcol[0]], 0, 3);
      __builtin_prefetch(&ga[(size_t)arow[0] * DIN + (i + 2) * BK + acol[0]], 0, 3);
    }

    // ---- fragments (ISA 7.12.2 wave32 layouts) + 16 WMMAs
    union Frag { v16bf v; u32x4 q[2]; } fa[4], fb;
#pragma unroll
    for (int ms = 0; ms < 4; ++ms) {   // A: lane = M; lanes<16: K 0-7 & 16-23, hi: 8-15 & 24-31
      const int m  = wm * 64 + ms * 16 + lane15;
      const int kb = laneHi * 8;
      fa[ms].q[0] = *(const u32x4*)&xsb[m * LSTR + kb];
      fa[ms].q[1] = *(const u32x4*)&xsb[m * LSTR + kb + 16];
    }
#pragma unroll
    for (int ns = 0; ns < 4; ++ns) {   // B: lane = N, K packed 2/VGPR; lanes<16: K 0-7 & 16-23, hi: 8-15 & 24-31
      const int n  = wn * 64 + ns * 16 + lane15;
      const int kb = laneHi * 8;
      fb.q[0] = *(const u32x4*)&asb[n * LSTR + kb];
      fb.q[1] = *(const u32x4*)&asb[n * LSTR + kb + 16];
#pragma unroll
      for (int ms = 0; ms < 4; ++ms)
        acc[ms][ns] = __builtin_amdgcn_wmma_f32_16x16x32_bf16(
            false, fa[ms].v, false, fb.v, (short)0, acc[ms][ns], false, false);
    }

    // ---- park next tile into the alternate LDS buffer; single barrier
    if (kn < DIN) {
      u16* xsn = &xs[(cur ^ 1) * BM * LSTR];
      u16* asn = &as[(cur ^ 1) * BN * LSTR];
#pragma unroll
      for (int q = 0; q < 2; ++q)
        *(u32x4*)&xsn[xrow[q] * LSTR + xcol[q]] = rx[q];
#pragma unroll
      for (int q = 0; q < 4; ++q)
        *(u32x4*)&asn[arow[q] * LSTR + acol[q]] = ra[q];
      __syncthreads();
    }
  }

  // ---- epilogue: C/D layout -> row-major out (lane = N, VGPR r -> M = r + 8*laneHi)
#pragma unroll
  for (int ms = 0; ms < 4; ++ms) {
#pragma unroll
    for (int ns = 0; ns < 4; ++ns) {
      const int n  = n0 + wn * 64 + ns * 16 + lane15;
      const int mb = m0 + wm * 64 + ms * 16 + laneHi * 8;
#pragma unroll
      for (int r = 0; r < 8; ++r)
        out[(size_t)(mb + r) * DOUT + n] = acc[ms][ns][r];
    }
  }
}

// ---------------------------------------------------------------------------
extern "C" void kernel_launch(void* const* d_in, const int* in_sizes, int n_in,
                              void* d_out, int out_size, void* d_ws, size_t ws_size,
                              hipStream_t stream) {
  (void)in_sizes; (void)n_in; (void)out_size; (void)ws_size;
  const float* x  = (const float*)d_in[0];
  const float* Wr = (const float*)d_in[1];
  const float* Wi = (const float*)d_in[2];
  float* out = (float*)d_out;

  char* ws = (char*)d_ws;
  float* w  = (float*)(ws);                                  // 256 KB taps
  u16*   xb = (u16*)(ws + (size_t)262144);                   // 32 MB bf16 x
  u16*   AT = (u16*)(ws + (size_t)262144 + 33554432);        // 32 MB bf16 operator

  k_make_w <<<256, 256, 0, stream>>>(Wr, Wi, w);
  k_make_AT<<<dim3(DOUT, KIN), 256, 0, stream>>>(w, AT);
  k_cvt_x  <<<(NB * (size_t)DIN) / (256 * 8), 256, 0, stream>>>(x, xb);
  k_gemm   <<<dim3(DOUT / BN, NB / BM), 256, 0, stream>>>(xb, AT, out);
}
